// Attention_89154931130941
// MI455X (gfx1250) — compile-verified, matrix operand order checked by reading, not run on hardware
//
#include <hip/hip_runtime.h>

// Problem constants (from reference)
#define B_   2
#define S_   2048
#define H_   1024
#define NH_  16
#define HD_  64
#define INV_SCALE 0.03125f      // 1/sqrt(H) = 1/32
#define EPS_ 1e-5f

typedef unsigned short u16;
typedef unsigned int   u32;
typedef __attribute__((ext_vector_type(16))) __bf16 v16bf;
typedef __attribute__((ext_vector_type(8)))  float  v8f;
typedef __attribute__((ext_vector_type(4)))  float  f32x4;
typedef __attribute__((ext_vector_type(2)))  u32    u32x2;
typedef __attribute__((ext_vector_type(4)))  u32    u32x4;
typedef __attribute__((ext_vector_type(4)))  int    v4i;

// ---- optional CDNA5 async global->LDS path (guarded; sync fallback) -------
#if defined(__has_builtin)
#if __has_builtin(__builtin_amdgcn_global_load_async_to_lds_b128) && \
    __has_builtin(__builtin_amdgcn_s_wait_asynccnt)
#define HAVE_ASYNC_LDS 1
#endif
#endif
#ifndef HAVE_ASYNC_LDS
#define HAVE_ASYNC_LDS 0
#endif

#if HAVE_ASYNC_LDS
typedef __attribute__((address_space(1))) v4i* g4p;   // prints as "int4 __device__ *"
typedef __attribute__((address_space(3))) v4i* l4p;   // prints as "int4 __shared__ *"
__device__ __forceinline__ void async_g2l_b128(const void* g, void* l) {
  // generic->AS1 is identity; generic LDS addr low 32 bits == LDS offset (ISA 10.2)
  g4p gp = (g4p)(unsigned long long)(uintptr_t)g;
  l4p lp = (l4p)(unsigned int)(uintptr_t)l;
  __builtin_amdgcn_global_load_async_to_lds_b128(gp, lp, 0, 0);
}
__device__ __forceinline__ void async_wait0() {
  __builtin_amdgcn_s_wait_asynccnt(0);
}
#endif

// fp32 -> bf16 round-to-nearest-even
__device__ __forceinline__ u16 f2bf(float f) {
  u32 u = __float_as_uint(f);
  return (u16)((u + 0x7FFFu + ((u >> 16) & 1u)) >> 16);
}

union Frag { v16bf v; u32x2 q[4]; };

// Gather a 16-lane-striped bf16 fragment (A or B operand) from a row whose
// K-dim is contiguous.  Per ISA layout: halves 0..3 -> K+0..3, 4..7 -> K+4..7,
// 8..11 -> K+16..19, 12..15 -> K+20..23; lanes 16..31 add +8 (koff0).
__device__ __forceinline__ v16bf ld_frag(const u16* p, int koff0) {
  Frag f;
  f.q[0] = *(const u32x2*)(p + koff0 + 0);
  f.q[1] = *(const u32x2*)(p + koff0 + 4);
  f.q[2] = *(const u32x2*)(p + koff0 + 16);
  f.q[3] = *(const u32x2*)(p + koff0 + 20);
  return f.v;
}

__device__ __forceinline__ v8f wmma_bf(v16bf a, v16bf b, v8f c) {
  return __builtin_amdgcn_wmma_f32_16x16x32_bf16(
      /*neg_a=*/false, a, /*neg_b=*/false, b,
      /*c_mod=*/(short)0, c, /*reuse_a=*/false, /*reuse_b=*/false);
}

// ---------------------------------------------------------------------------
// GEMM:  out[M,N] = A[M,K] * W[K,N] + bias  (+ residual, f32 out for FC path)
// Block tile 32x256, 8 waves, each wave 16x64 (4 accumulators), BK=32.
// ---------------------------------------------------------------------------
#define BM 32
#define BN 256
#define BK 32
#define AP 40   // LDS pitch for A tile (halves)
#define BP 40   // LDS pitch for transposed W tile (halves)

template<bool A_BF16, bool WRITE_F32>
__global__ __launch_bounds__(256)
void gemm_bias_kernel(const void* __restrict__ Ap, const float* __restrict__ W,
                      const float* __restrict__ bias, const float* __restrict__ residual,
                      u16* __restrict__ out_bf, float* __restrict__ out_f,
                      int M, int N, int K)
{
  __shared__ u16 As[BM * AP];
  __shared__ u16 Bs[BN * BP];   // transposed: [n][k] so K is contiguous

  const int tid   = threadIdx.x;
  const int lane  = tid & 31;
  const int wid   = tid >> 5;
  const int wm    = wid >> 2;        // 0..1 : wave row
  const int wn    = wid & 3;         // 0..3 : wave col
  const int hi    = lane >> 4;
  const int lm    = lane & 15;
  const int koff0 = hi * 8;

  const int m0 = blockIdx.y * BM;
  const int n0 = blockIdx.x * BN;

  v8f acc[4] = {};

  for (int k0 = 0; k0 < K; k0 += BK) {
    __syncthreads();
    // ---- stage A tile (32x32) ----
    if (A_BF16) {
#if HAVE_ASYNC_LDS
      if (tid < 128) {                       // 128 x b128 = 1024 bf16
        int r = tid >> 2, c = (tid & 3) * 8;
        async_g2l_b128(((const u16*)Ap) + (size_t)(m0 + r) * K + k0 + c,
                       &As[r * AP + c]);
      }
#else
      {
        int r = tid >> 3, c = (tid & 7) * 4;  // 4 bf16 per thread
        u32x2 d = *(const u32x2*)(((const u16*)Ap) + (size_t)(m0 + r) * K + k0 + c);
        *(u32x2*)(&As[r * AP + c]) = d;
      }
#endif
    } else {
      int r = tid >> 3, c = (tid & 7) * 4;    // float4 load, 4x bf16 store
      f32x4 a4 = *(const f32x4*)(((const float*)Ap) + (size_t)(m0 + r) * K + k0 + c);
      u16 h[4] = { f2bf(a4.x), f2bf(a4.y), f2bf(a4.z), f2bf(a4.w) };
      *(u32x2*)(&As[r * AP + c]) = *(const u32x2*)h;
    }
    // ---- stage W tile transposed ([k][n] global -> [n][k] LDS) ----
#pragma unroll
    for (int i = 0; i < 8; ++i) {             // 8 x float4 per thread
      int g = tid + i * 256;
      int r = g >> 6;                         // k 0..31
      int c = (g & 63) * 4;                   // n
      f32x4 w4 = *(const f32x4*)(&W[(size_t)(k0 + r) * N + n0 + c]);
      Bs[(c + 0) * BP + r] = f2bf(w4.x);
      Bs[(c + 1) * BP + r] = f2bf(w4.y);
      Bs[(c + 2) * BP + r] = f2bf(w4.z);
      Bs[(c + 3) * BP + r] = f2bf(w4.w);
    }
    // prefetch next W K-chunk into cache (global_prefetch_b8)
    if (k0 + BK < K) {
      int pr = tid >> 3;
      int pc = (tid & 7) * 32;
      __builtin_prefetch(&W[(size_t)(k0 + BK + pr) * N + n0 + pc], 0, 0);
    }
#if HAVE_ASYNC_LDS
    if (A_BF16) async_wait0();
#endif
    __syncthreads();

    v16bf afrag = ld_frag(&As[(wm * 16 + lm) * AP], koff0);
#pragma unroll
    for (int s = 0; s < 4; ++s) {
      v16bf bfrag = ld_frag(&Bs[(wn * 64 + s * 16 + lm) * BP], koff0);
      acc[s] = wmma_bf(afrag, bfrag, acc[s]);
    }
  }

  // epilogue: C/D layout: col = lane%16, row = r + 8*(lane>=16)
#pragma unroll
  for (int s = 0; s < 4; ++s) {
    int n = n0 + wn * 64 + s * 16 + lm;
    float bv = bias[n];
#pragma unroll
    for (int r = 0; r < 8; ++r) {
      int m = m0 + wm * 16 + r + hi * 8;
      float val = acc[s][r] + bv;
      if (WRITE_F32) {
        out_f[(size_t)m * N + n] = val + residual[(size_t)m * N + n];
      } else {
        out_bf[(size_t)m * N + n] = f2bf(val);
      }
    }
  }
}

// ---------------------------------------------------------------------------
// Flash attention: one workgroup = (batch, head, 128-query block), one wave
// per 16-query tile.  K/V tiles (32 keys) staged in LDS, online softmax.
// ---------------------------------------------------------------------------
#define KP 72   // K tile pitch:  [key][hd], hd contiguous (64+8)
#define VP 40   // V tile pitch:  [hd][key], key contiguous (32+8)
#define PP 40   // per-wave P scratch pitch

__global__ __launch_bounds__(256)
void flash_attn_kernel(const u16* __restrict__ qh, const u16* __restrict__ kh,
                       const u16* __restrict__ vh, const int* __restrict__ mask,
                       u16* __restrict__ out)
{
  __shared__ u16 Ks[32 * KP];
  __shared__ u16 Vs[HD_ * VP];
  __shared__ u16 Ps[8 * 16 * PP];

  const int tid   = threadIdx.x;
  const int lane  = tid & 31;
  const int wid   = tid >> 5;
  const int hi    = lane >> 4;
  const int lm    = lane & 15;
  const int koff0 = hi * 8;

  const int b  = blockIdx.z;
  const int h  = blockIdx.y;
  const int q0 = blockIdx.x * 128 + wid * 16;

  // Q fragments, 2 chunks of K=32 over HD=64, held in registers for all keys
  const u16* qbase = qh + ((size_t)(b * S_) + q0 + lm) * H_ + h * HD_;
  v16bf qf0 = ld_frag(qbase, koff0);
  v16bf qf1 = ld_frag(qbase + 32, koff0);

  float rmax[8], rsum[8];
  v8f O[4] = {};
#pragma unroll
  for (int r = 0; r < 8; ++r) { rmax[r] = -1e30f; rsum[r] = 0.f; }

  u16* Pw = &Ps[wid * 16 * PP];

  for (int kt = 0; kt < S_; kt += 32) {
    __syncthreads();
    // ---- stage K tile: 32 keys x 64 hd (b128 per thread) ----
    {
      int r = tid >> 3;              // key
      int c = (tid & 7) * 8;         // hd
      const u16* g = kh + ((size_t)(b * S_) + kt + r) * H_ + h * HD_ + c;
#if HAVE_ASYNC_LDS
      async_g2l_b128(g, &Ks[r * KP + c]);
#else
      *(u32x4*)(&Ks[r * KP + c]) = *(const u32x4*)g;
#endif
    }
    // ---- stage V tile transposed -> [hd][key] ----
#pragma unroll
    for (int i = 0; i < 8; ++i) {
      int idx = tid + i * 256;
      int r = idx >> 6;              // key
      int c = idx & 63;              // hd
      Vs[c * VP + r] = vh[((size_t)(b * S_) + kt + r) * H_ + h * HD_ + c];
    }
#if HAVE_ASYNC_LDS
    async_wait0();
#endif
    __syncthreads();

    // scores: two 16x16 tiles (N = 16 keys each), K-dim = HD = 64
    v8f e[2];
#pragma unroll
    for (int s = 0; s < 2; ++s) {
      v16bf kf0 = ld_frag(&Ks[(s * 16 + lm) * KP], koff0);
      v16bf kf1 = ld_frag(&Ks[(s * 16 + lm) * KP + 32], koff0);
      v8f c = {};
      c = wmma_bf(qf0, kf0, c);
      c = wmma_bf(qf1, kf1, c);
      e[s] = c;
    }

    // mask + scale (reference: mask->-1e9, then /sqrt(H))
    float sc[2][8];
#pragma unroll
    for (int s = 0; s < 2; ++s) {
      int col = kt + s * 16 + lm;
#pragma unroll
      for (int r = 0; r < 8; ++r) {
        int row = q0 + r + hi * 8;
        int mv = mask[((size_t)(b * S_) + row) * S_ + col];
        sc[s][r] = mv ? e[s][r] * INV_SCALE : (-1e9f * INV_SCALE);
      }
    }

    // online softmax (row values live across 16 lanes of each half-wave)
    float alpha[8];
#pragma unroll
    for (int r = 0; r < 8; ++r) {
      float mt = fmaxf(sc[0][r], sc[1][r]);
#pragma unroll
      for (int off = 8; off >= 1; off >>= 1)
        mt = fmaxf(mt, __shfl_xor(mt, off, 32));
      float mnew = fmaxf(rmax[r], mt);
      alpha[r] = __expf(rmax[r] - mnew);
      rmax[r] = mnew;
      float p0 = __expf(sc[0][r] - mnew);
      float p1 = __expf(sc[1][r] - mnew);
      sc[0][r] = p0; sc[1][r] = p1;
      float part = p0 + p1;
#pragma unroll
      for (int off = 8; off >= 1; off >>= 1)
        part += __shfl_xor(part, off, 32);
      rsum[r] = rsum[r] * alpha[r] + part;
    }

    // rescale running output
#pragma unroll
    for (int s = 0; s < 4; ++s)
#pragma unroll
      for (int r = 0; r < 8; ++r)
        O[s][r] *= alpha[r];

    // C-layout P -> per-wave LDS -> A-fragment layout (wave-local round trip)
#pragma unroll
    for (int s = 0; s < 2; ++s)
#pragma unroll
      for (int r = 0; r < 8; ++r)
        Pw[(r + hi * 8) * PP + s * 16 + lm] = f2bf(sc[s][r]);

    v16bf pf = ld_frag(&Pw[lm * PP], koff0);
#pragma unroll
    for (int s = 0; s < 4; ++s) {
      v16bf vf = ld_frag(&Vs[(s * 16 + lm) * VP], koff0);
      O[s] = wmma_bf(pf, vf, O[s]);   // 16x16x32: K = 32 keys
    }
  }

  // normalize and write bf16 attention output
#pragma unroll
  for (int s = 0; s < 4; ++s) {
#pragma unroll
    for (int r = 0; r < 8; ++r) {
      int row = q0 + r + hi * 8;
      int col = h * HD_ + s * 16 + lm;
      out[((size_t)(b * S_) + row) * H_ + col] = f2bf(O[s][r] / rsum[r]);
    }
  }
}

// ---------------------------------------------------------------------------
// LayerNorm over H=1024, one block per row.
// ---------------------------------------------------------------------------
__global__ __launch_bounds__(256)
void layernorm_kernel(const float* __restrict__ y, const float* __restrict__ gamma,
                      const float* __restrict__ beta, float* __restrict__ out)
{
  __shared__ float red[16];
  const int row = blockIdx.x;
  const int tid = threadIdx.x;
  const float* yr = y + (size_t)row * H_;

  float v[4];
  float s = 0.f, ss = 0.f;
#pragma unroll
  for (int i = 0; i < 4; ++i) {
    v[i] = yr[tid + i * 256];
    s += v[i]; ss += v[i] * v[i];
  }
#pragma unroll
  for (int off = 16; off >= 1; off >>= 1) {
    s  += __shfl_xor(s,  off, 32);
    ss += __shfl_xor(ss, off, 32);
  }
  const int wid = tid >> 5, lane = tid & 31;
  if (lane == 0) { red[wid] = s; red[8 + wid] = ss; }
  __syncthreads();
  if (tid == 0) {
    float a = 0.f, b2 = 0.f;
    for (int i = 0; i < 8; ++i) { a += red[i]; b2 += red[8 + i]; }
    red[0] = a; red[8] = b2;
  }
  __syncthreads();
  const float mu   = red[0] * (1.f / H_);
  const float var  = red[8] * (1.f / H_) - mu * mu;
  const float rstd = rsqrtf(var + EPS_);
#pragma unroll
  for (int i = 0; i < 4; ++i) {
    int c = tid + i * 256;
    out[(size_t)row * H_ + c] = gamma[c] * (v[i] - mu) * rstd + beta[c];
  }
}

// ---------------------------------------------------------------------------
extern "C" void kernel_launch(void* const* d_in, const int* in_sizes, int n_in,
                              void* d_out, int out_size, void* d_ws, size_t ws_size,
                              hipStream_t stream)
{
  (void)in_sizes; (void)n_in; (void)out_size; (void)ws_size;
  const float* q    = (const float*)d_in[0];
  const float* k    = (const float*)d_in[1];
  const float* v    = (const float*)d_in[2];
  const int*   mask = (const int*)d_in[3];
  const float* Wq   = (const float*)d_in[4];
  const float* bq   = (const float*)d_in[5];
  const float* Wk   = (const float*)d_in[6];
  const float* bk   = (const float*)d_in[7];
  const float* Wv   = (const float*)d_in[8];
  const float* bv   = (const float*)d_in[9];
  const float* Wfc  = (const float*)d_in[10];
  const float* bfc  = (const float*)d_in[11];
  const float* gamma = (const float*)d_in[12];
  const float* beta  = (const float*)d_in[13];

  const size_t MN = (size_t)B_ * S_ * H_;   // 4M elements
  char* ws = (char*)d_ws;
  u16*   qh = (u16*)(ws);
  u16*   kh = (u16*)(ws + MN * 2);
  u16*   vh = (u16*)(ws + MN * 4);
  u16*   ao = (u16*)(ws + MN * 6);
  float* y  = (float*)(ws + MN * 8);

  const int M = B_ * S_;
  dim3 gg(H_ / BN, M / BM);   // (4, 128)

  gemm_bias_kernel<false, false><<<gg, 256, 0, stream>>>(q, Wq, bq, nullptr, qh, nullptr, M, H_, H_);
  gemm_bias_kernel<false, false><<<gg, 256, 0, stream>>>(k, Wk, bk, nullptr, kh, nullptr, M, H_, H_);
  gemm_bias_kernel<false, false><<<gg, 256, 0, stream>>>(v, Wv, bv, nullptr, vh, nullptr, M, H_, H_);

  flash_attn_kernel<<<dim3(S_ / 128, NH_, B_), 256, 0, stream>>>(qh, kh, vh, mask, ao);

  gemm_bias_kernel<true, true><<<gg, 256, 0, stream>>>(ao, Wfc, bfc, q, nullptr, y, M, H_, H_);

  layernorm_kernel<<<M, 256, 0, stream>>>(y, gamma, beta, (float*)d_out);
}
